// VARGG_model_44899588112569
// MI455X (gfx1250) — compile-verified, matrix operand order checked by reading, not run on hardware
//
#include <hip/hip_runtime.h>
#include <math.h>

typedef __attribute__((ext_vector_type(16))) _Float16 v16h;
typedef __attribute__((ext_vector_type(8)))  _Float16 v8h;
typedef __attribute__((ext_vector_type(8)))  float    v8f;
typedef __attribute__((ext_vector_type(4)))  int      v4i;

#define NND   4096     // N_NODES
#define NEDG  65536    // N_EDGES
#define INDIM 3000     // IN_DIM

#if __has_builtin(__builtin_amdgcn_global_load_async_to_lds_b128) && \
    __has_builtin(__builtin_amdgcn_s_wait_asynccnt)
#define ASYNC_LDS 1
typedef __attribute__((address_space(1))) v4i* gptr_b128;   // global 16B chunk
typedef __attribute__((address_space(3))) v4i* lptr_b128;   // LDS 16B chunk
#else
#define ASYNC_LDS 0
#endif

// ---------------------------------------------------------------------------
// f16-operand WMMA GEMM:  out = epilogue( alpha * (A[M,K] x B) + bias )
//   BT=1: B is W[N,K] row-major  (computes A @ W^T)
//   BT=0: B is B[K,N] row-major  (computes A @ B)
// Outputs: optional f32 C (ldc) and/or f16 Ch (ldch); Ch column index can be
// remapped into a group-padded layout: cc = (col/cgrp)*cgrpPad + col%cgrp.
// Tile 64x64, 256 threads (8 waves).
// BT=1 path uses GLOBAL_LOAD_ASYNC_TO_LDS_B128 with double-buffered LDS
// (ASYNCcnt-tracked DMA, no VGPR staging); ragged K-tail and BT=0 use a
// register-pipelined path. All pointers/strides 16B aligned by construction.
// ---------------------------------------------------------------------------
template <int BT>
__global__ __launch_bounds__(256)
void gemm_wmma_h(const _Float16* __restrict__ A, int lda,
                 const _Float16* __restrict__ B, int ldb,
                 float* __restrict__ C, int ldc,
                 _Float16* __restrict__ Ch, int ldch, int cgrp, int cgrpPad,
                 int M, int N, int K,
                 const float* __restrict__ bias,
                 const float* __restrict__ scale,
                 const float* __restrict__ shift,
                 float alpha, int act)
{
    __shared__ _Float16 As[2][64][40];   // row stride 80B -> 16B-aligned 8-chunks
    __shared__ _Float16 Bs[2][64][40];

    const int tid  = threadIdx.x;
    const int lane = tid & 31;
    const int wave = tid >> 5;
    const int m0 = blockIdx.y * 64;
    const int n0 = blockIdx.x * 64;
    const int wm = (wave >> 1) * 16;
    const int wn = (wave & 1) * 32;
    const int lr = lane & 15;
    const int lh = lane >> 4;

    // load mapping: thread -> row r, 8-wide k-chunk c8
    const int r  = tid >> 2;         // 0..63
    const int c8 = (tid & 3) << 3;   // 0,8,16,24
    // BT=0 mapping (B[K,N], contiguous along N): thread -> k row rk, n-chunk nc
    const int rk = tid >> 3;         // 0..31
    const int nc = (tid & 7) << 3;   // 0..56

    // clamp rows (OOB output rows/cols are never stored, so no zero-mask needed)
    const int gmA = (m0 + r < M) ? (m0 + r) : (M - 1);
    const int gnB = (n0 + r < N) ? (n0 + r) : (N - 1);

    const _Float16* pa_base  = A + (size_t)gmA * lda + c8;
    const _Float16* pbT_base = BT ? (B + (size_t)gnB * ldb + c8) : B;

    v8h arv, brv;
    v8f acc0 = {};
    v8f acc1 = {};

    auto loadMain = [&](int k0) {   // whole 32-wide k-slab in range
        arv = *(const v8h*)(pa_base + k0);
        __builtin_prefetch(pa_base + k0 + 64, 0, 1);
        if (BT) {
            brv = *(const v8h*)(pbT_base + k0);
        } else {
            brv = *(const v8h*)(B + (size_t)(k0 + rk) * ldb + n0 + nc);
        }
    };
    auto loadTailRegs = [&](int k0, int rem) {  // rem in 1..31; zero-mask k-pad
        #pragma unroll
        for (int j = 0; j < 8; ++j) {
            int kk = c8 + j;
            int ki = (kk < rem) ? kk : 0;
            _Float16 v = pa_base[k0 + ki];
            arv[j] = (kk < rem) ? v : (_Float16)0.0f;
        }
        if (BT) {
            #pragma unroll
            for (int j = 0; j < 8; ++j) {
                int kk = c8 + j;
                int ki = (kk < rem) ? kk : 0;
                _Float16 v = pbT_base[k0 + ki];
                brv[j] = (kk < rem) ? v : (_Float16)0.0f;
            }
        } else {
            int kr = (rk < rem) ? (k0 + rk) : k0;
            const _Float16* pb = B + (size_t)kr * ldb + n0 + nc;
            #pragma unroll
            for (int j = 0; j < 8; ++j) {
                _Float16 v = pb[j];
                brv[j] = (rk < rem) ? v : (_Float16)0.0f;
            }
        }
    };
    auto storeRegs = [&](int buf) {
        *(v8h*)(&As[buf][r][c8]) = arv;                 // ds_store_b128
        if (BT) {
            *(v8h*)(&Bs[buf][r][c8]) = brv;             // ds_store_b128
        } else {
            #pragma unroll
            for (int j = 0; j < 8; ++j) Bs[buf][nc + j][rk] = brv[j];
        }
    };
    auto computeTile = [&](int buf) {
        // fragments per ISA 16-bit 16x32 layout:
        // lane half 0 holds K {0-7,16-23}, half 1 holds K {8-15,24-31}
        v16h a, b0, b1;
        #pragma unroll
        for (int j = 0; j < 16; ++j) {
            int kk = ((j >> 3) << 4) + lh * 8 + ((j >> 1) & 3) * 2 + (j & 1);
            a[j]  = As[buf][wm + lr][kk];
            b0[j] = Bs[buf][wn + lr][kk];
            b1[j] = Bs[buf][wn + 16 + lr][kk];
        }
        acc0 = __builtin_amdgcn_wmma_f32_16x16x32_f16(false, a, false, b0,
                                                      (short)0, acc0, false, false);
        acc1 = __builtin_amdgcn_wmma_f32_16x16x32_f16(false, a, false, b1,
                                                      (short)0, acc1, false, false);
    };

    const int kfull = K & ~31;

#if ASYNC_LDS
    if (BT) {
        // -------- async DMA pipeline (double-buffered LDS) --------
        const int ntMain = kfull >> 5;
        auto issueAsync = [&](int ti) {
            int buf = ti & 1;
            __builtin_amdgcn_global_load_async_to_lds_b128(
                (gptr_b128)(pa_base + ti * 32),
                (lptr_b128)(&As[buf][r][c8]), 0, 0);
            __builtin_amdgcn_global_load_async_to_lds_b128(
                (gptr_b128)(pbT_base + ti * 32),
                (lptr_b128)(&Bs[buf][r][c8]), 0, 0);
        };
        if (ntMain) issueAsync(0);
        for (int ti = 0; ti < ntMain; ++ti) {
            if (ti + 1 < ntMain) {
                issueAsync(ti + 1);                       // next tile in flight
                __builtin_amdgcn_s_wait_asynccnt(2);      // drain current tile only
            } else {
                __builtin_amdgcn_s_wait_asynccnt(0);
            }
            __syncthreads();                               // tile visible to all waves
            computeTile(ti & 1);
            __syncthreads();                               // reads done before rewrite
        }
        if (kfull < K) {                                   // ragged tail: register path
            loadTailRegs(kfull, K - kfull);
            storeRegs(ntMain & 1);
            __syncthreads();
            computeTile(ntMain & 1);
        }
    } else
#endif
    {
        // -------- register-pipelined path --------
        if (kfull > 0) loadMain(0); else loadTailRegs(0, K);
        int t = 0;
        for (int k0 = 0; k0 < K; k0 += 32, ++t) {
            if (t) __syncthreads();        // previous compute done -> LDS reusable
            storeRegs(0);
            int k1 = k0 + 32;
            if (k1 < kfull)  loadMain(k1); // issue next tile's loads early
            else if (k1 < K) loadTailRegs(k1, K - k1);
            __syncthreads();
            computeTile(0);
        }
    }

    // C/D layout: VGPR i -> row wm + lh*8 + i ; col = lane&15
    const int rowb = m0 + wm + lh * 8;
    #pragma unroll
    for (int sub = 0; sub < 2; ++sub) {
        v8f acc = sub ? acc1 : acc0;
        int col = n0 + wn + sub * 16 + lr;
        if (col < N) {
            float bb = bias  ? bias[col]  : 0.0f;
            float sc = scale ? scale[col] : 1.0f;
            float sh = shift ? shift[col] : 0.0f;
            int cc = col;
            if (cgrp) { int g = col / cgrp; cc = g * cgrpPad + (col - g * cgrp); }
            #pragma unroll
            for (int i = 0; i < 8; ++i) {
                int rr = rowb + i;
                if (rr < M) {
                    float v = acc[i] * alpha + bb;
                    v = v * sc + sh;
                    if (act) v = (v > 0.0f) ? v : (__expf(v) - 1.0f);  // ELU
                    if (C)  C[(size_t)rr * ldc + col] = v;
                    if (Ch) Ch[(size_t)rr * ldch + cc] = (_Float16)v;
                }
            }
        }
    }
}

// ---------------------------------------------------------------------------
// Row softmax: reads f32 scores, writes normalized f16 probabilities.
// ---------------------------------------------------------------------------
__global__ __launch_bounds__(256)
void softmax_rows(float* __restrict__ S, _Float16* __restrict__ O16, int Ncols)
{
    const int row  = blockIdx.x;
    float* p = S + (size_t)row * Ncols;
    _Float16* o = O16 + (size_t)row * Ncols;
    const int lane = threadIdx.x & 31;
    const int wv   = threadIdx.x >> 5;
    __shared__ float redm[8];
    __shared__ float reds[8];

    float m = -3.4e38f;
    for (int i = threadIdx.x; i < Ncols; i += 256) m = fmaxf(m, p[i]);
    for (int o2 = 16; o2; o2 >>= 1) m = fmaxf(m, __shfl_xor(m, o2, 32));
    if (!lane) redm[wv] = m;
    __syncthreads();
    m = redm[0];
    #pragma unroll
    for (int i = 1; i < 8; ++i) m = fmaxf(m, redm[i]);

    float s = 0.0f;
    for (int i = threadIdx.x; i < Ncols; i += 256) {
        float e = __expf(p[i] - m);
        p[i] = e;
        s += e;
    }
    for (int o2 = 16; o2; o2 >>= 1) s += __shfl_xor(s, o2, 32);
    if (!lane) reds[wv] = s;
    __syncthreads();
    s = 0.0f;
    #pragma unroll
    for (int i = 0; i < 8; ++i) s += reds[i];
    float inv = 1.0f / s;
    for (int i = threadIdx.x; i < Ncols; i += 256)
        o[i] = (_Float16)(p[i] * inv);
}

// ---------------------------------------------------------------------------
// ResGatedGraphConv edge aggregation:
//   agg[dst] += sigmoid(K[dst]+Q[src]) * V[src]   (agg pre-seeded with Ws x + bs)
// ---------------------------------------------------------------------------
__global__ void rg_edges(const float* __restrict__ Kb, const float* __restrict__ Qb,
                         const float* __restrict__ Vb,
                         const int* __restrict__ src, const int* __restrict__ dst,
                         float* __restrict__ agg, int E, int D)
{
    int i = blockIdx.x * blockDim.x + threadIdx.x;
    if (i >= E * D) return;
    int e = i / D, d = i - e * D;
    int s = src[e], t = dst[e];
    float g = Kb[(size_t)t * D + d] + Qb[(size_t)s * D + d];
    g = 1.0f / (1.0f + __expf(-g));
    atomicAdd(&agg[(size_t)t * D + d], g * Vb[(size_t)s * D + d]);
}

// y16 = elu(x * scale[c] + shift[c])
__global__ void bn_elu_h(const float* __restrict__ in, _Float16* __restrict__ o,
                         const float* __restrict__ sc, const float* __restrict__ sh,
                         int n, int D)
{
    int i = blockIdx.x * blockDim.x + threadIdx.x;
    if (i >= n) return;
    int c = i % D;
    float v = in[i] * sc[c] + sh[c];
    v = (v > 0.0f) ? v : (__expf(v) - 1.0f);
    o[i] = (_Float16)v;
}

// strided column-block copy: dst[r*ldd + c] = src[r*lds + c]
__global__ void copy_cols(float* __restrict__ dst, int ldd,
                          const float* __restrict__ src, int lds,
                          int rows, int cols)
{
    int i = blockIdx.x * blockDim.x + threadIdx.x;
    if (i >= rows * cols) return;
    int r = i / cols, c = i - r * cols;
    dst[(size_t)r * ldd + c] = src[(size_t)r * lds + c];
}

// plain f32 -> f16 convert
__global__ void cvt_f16(const float* __restrict__ s, _Float16* __restrict__ d, int n)
{
    int i = blockIdx.x * blockDim.x + threadIdx.x;
    if (i < n) d[i] = (_Float16)s[i];
}

// f32 [rows,K] -> f16 [rows,Kp] with per-group K padding (grp -> grpPad, zero pad)
__global__ void cvt_f16_padK(const float* __restrict__ s, _Float16* __restrict__ d,
                             int rows, int K, int Kp, int grp, int grpPad)
{
    int i = blockIdx.x * blockDim.x + threadIdx.x;
    if (i >= rows * Kp) return;
    int r = i / Kp, c = i - r * Kp;
    int g = c / grpPad, pos = c - g * grpPad;
    float v = (pos < grp) ? s[(size_t)r * K + g * grp + pos] : 0.0f;
    d[i] = (_Float16)v;
}

__global__ void fill_h(_Float16* __restrict__ d, int n)
{
    int i = blockIdx.x * blockDim.x + threadIdx.x;
    if (i < n) d[i] = (_Float16)0.0f;
}

// Student-t soft assignment: q[n,k] ~ (1 + |z-c|^2 + 1e-8)^-1, row-normalized (alpha=1)
__global__ void cluster_q_k(const float* __restrict__ z, const float* __restrict__ cl,
                            float* __restrict__ q, int N)
{
    int n = blockIdx.x * blockDim.x + threadIdx.x;
    if (n >= N) return;
    float qv[15];
    float s = 0.0f;
    #pragma unroll
    for (int k = 0; k < 15; ++k) {
        float d2 = 0.0f;
        #pragma unroll
        for (int j = 0; j < 48; ++j) {
            float d = z[(size_t)n * 48 + j] - cl[k * 48 + j];
            d2 += d * d;
        }
        float t = 1.0f / (1.0f + d2 + 1e-8f);
        qv[k] = t;
        s += t;
    }
    float inv = 1.0f / s;
    #pragma unroll
    for (int k = 0; k < 15; ++k) q[(size_t)n * 15 + k] = qv[k] * inv;
}

// ---------------------------------------------------------------------------
extern "C" void kernel_launch(void* const* d_in, const int* in_sizes, int n_in,
                              void* d_out, int out_size, void* d_ws, size_t ws_size,
                              hipStream_t stream)
{
    (void)in_sizes; (void)n_in; (void)out_size; (void)ws_size;
    auto F = [&](int i) { return (const float*)d_in[i]; };
    const float* x   = F(0);
    const int* ei    = (const int*)d_in[1];
    const int* esrc  = ei;
    const int* edst  = ei + NEDG;

    float* out = (float*)d_out;

    // ---- bump allocator over workspace (16B aligned) ----
    char* wsb = (char*)d_ws;
    size_t off = 0;
    auto allocH = [&](size_t n) {
        _Float16* p = (_Float16*)(wsb + off);
        off = (off + n * 2 + 15) & ~(size_t)15;
        return p;
    };
    auto allocF = [&](size_t n) {
        float* p = (float*)(wsb + off);
        off = (off + n * 4 + 15) & ~(size_t)15;
        return p;
    };

    // f16 weight/activation buffers
    _Float16* xh    = allocH((size_t)NND * INDIM);
    _Float16* a0i_h = allocH((size_t)9000 * INDIM);
    _Float16* a0o_h = allocH((size_t)INDIM * 3072);   // padded-K layout (375->384)
    _Float16* e0w_h = allocH((size_t)64 * INDIM);
    _Float16* a1i_h = allocH(192 * 64);
    _Float16* a1o_h = allocH(64 * 64);
    _Float16* e1w_h = allocH(40 * 64);
    _Float16* d0w_h = allocH(64 * 48);
    _Float16* d1w_h = allocH((size_t)INDIM * 64);
    _Float16* qkv16 = allocH((size_t)NND * 9216);     // head-padded Q|K|V
    float*    S32   = allocF((size_t)NND * NND);
    _Float16* S16   = allocH((size_t)NND * NND);
    _Float16* att16 = allocH((size_t)NND * 3072);     // head-padded attn out
    _Float16* h16   = allocH((size_t)NND * INDIM);
    _Float16* h64h  = allocH((size_t)NND * 64);
    _Float16* qkv1h = allocH((size_t)NND * 192);
    _Float16* at1h  = allocH((size_t)NND * 64);
    _Float16* h64bh = allocH((size_t)NND * 64);
    _Float16* feat16= allocH((size_t)NND * 40);
    _Float16* cb1h  = allocH((size_t)NND * 80);
    _Float16* cb2h  = allocH((size_t)NND * 80);
    _Float16* z16   = allocH((size_t)NND * 48);
    _Float16* dbufh = allocH((size_t)NND * 64);
    float* kb   = allocF((size_t)NND * 80);
    float* qb   = allocF((size_t)NND * 80);
    float* vb   = allocF((size_t)NND * 80);
    float* aggb = allocF((size_t)NND * 80);

    // ---- output region offsets (tuple order) ----
    const size_t OZ  = 0;
    const size_t OMU = (size_t)NND * 48;
    const size_t OLV = OMU + (size_t)NND * 8;
    const size_t ODE = OLV + (size_t)NND * 8;
    const size_t OQ  = ODE + (size_t)NND * INDIM;
    const size_t OFX = OQ  + (size_t)NND * 15;
    const size_t OGZ = OFX + (size_t)NND * 40;

    auto cvt = [&](const float* s, _Float16* d, int n) {
        cvt_f16<<<(n + 255) / 256, 256, 0, stream>>>(s, d, n);
    };
    auto gemm = [&](const _Float16* A, int lda, const _Float16* B, int ldb, int bT,
                    float* C, int ldc, _Float16* Ch, int ldch, int cgrp, int cgrpPad,
                    int M, int N, int K,
                    const float* bias, const float* sc, const float* sh,
                    float alpha, int act) {
        dim3 g((unsigned)((N + 63) / 64), (unsigned)((M + 63) / 64));
        if (bT)
            gemm_wmma_h<1><<<g, 256, 0, stream>>>(A, lda, B, ldb, C, ldc, Ch, ldch,
                                                  cgrp, cgrpPad, M, N, K,
                                                  bias, sc, sh, alpha, act);
        else
            gemm_wmma_h<0><<<g, 256, 0, stream>>>(A, lda, B, ldb, C, ldc, Ch, ldch,
                                                  cgrp, cgrpPad, M, N, K,
                                                  bias, sc, sh, alpha, act);
    };

    // ===== one-time f16 conversion of x and all weights =====
    cvt(x,     xh,    NND * INDIM);
    cvt(F(2),  a0i_h, 9000 * INDIM);
    {   // a0_out -> padded-K layout [3000][3072], zero pads
        int tot = INDIM * 3072;
        cvt_f16_padK<<<(tot + 255) / 256, 256, 0, stream>>>(F(3), a0o_h, INDIM,
                                                            INDIM, 3072, 375, 384);
    }
    cvt(F(4),  e0w_h, 64 * INDIM);
    cvt(F(8),  a1i_h, 192 * 64);
    cvt(F(9),  a1o_h, 64 * 64);
    cvt(F(10), e1w_h, 40 * 64);
    cvt(F(70), d0w_h, 64 * 48);
    cvt(F(74), d1w_h, INDIM * 64);

    const int pb[6]  = {14, 24, 34, 44, 54, 62};
    const int din[6] = {40, 80, 64, 32, 16, 16};
    const int dou[6] = {80, 64, 32, 16,  8,  8};
    _Float16* rgw[6][4];
    for (int l = 0; l < 6; ++l)
        for (int w = 0; w < 4; ++w) {
            int n = dou[l] * din[l];
            rgw[l][w] = allocH(n);
            cvt(F(pb[l] + 2 * w), rgw[l][w], n);
        }

    // ===== MHA over raw input (E=3000, 8 heads, hd=375; head-padded to 384) =====
    gemm(xh, INDIM, a0i_h, INDIM, 1, nullptr, 0, qkv16, 9216, 375, 384,
         NND, 9000, INDIM, nullptr, nullptr, nullptr, 1.0f, 0);
    fill_h<<<(NND * 3072 + 255) / 256, 256, 0, stream>>>(att16, NND * 3072);
    const float inv375 = 1.0f / sqrtf(375.0f);
    for (int h = 0; h < 8; ++h) {
        const _Float16* Qh = qkv16 + h * 384;
        const _Float16* Kh = qkv16 + 3072 + h * 384;
        const _Float16* Vh = qkv16 + 6144 + h * 384;
        gemm(Qh, 9216, Kh, 9216, 1, S32, NND, nullptr, 0, 0, 0,
             NND, NND, 375, nullptr, nullptr, nullptr, inv375, 0);
        softmax_rows<<<NND, 256, 0, stream>>>(S32, S16, NND);
        gemm(S16, NND, Vh, 9216, 0, nullptr, 0, att16 + h * 384, 3072, 0, 0,
             NND, 375, NND, nullptr, nullptr, nullptr, 1.0f, 0);
    }
    // output projection over padded K=3072 (pads zeroed in both operands)
    gemm(att16, 3072, a0o_h, 3072, 1, nullptr, 0, h16, INDIM, 0, 0,
         NND, INDIM, 3072, nullptr, nullptr, nullptr, 1.0f, 0);

    // ===== encoder stage 0: Linear(3000->64) + BN + ELU =====
    gemm(h16, INDIM, e0w_h, INDIM, 1, nullptr, 0, h64h, 64, 0, 0,
         NND, 64, INDIM, F(5), F(6), F(7), 1.0f, 1);

    // ===== MHA over 64-dim features (8 heads, hd=8) =====
    gemm(h64h, 64, a1i_h, 64, 1, nullptr, 0, qkv1h, 192, 0, 0,
         NND, 192, 64, nullptr, nullptr, nullptr, 1.0f, 0);
    const float inv8 = 1.0f / sqrtf(8.0f);
    for (int h = 0; h < 8; ++h) {
        const _Float16* Qh = qkv1h + h * 8;
        const _Float16* Kh = qkv1h + 64 + h * 8;
        const _Float16* Vh = qkv1h + 128 + h * 8;
        gemm(Qh, 192, Kh, 192, 1, S32, NND, nullptr, 0, 0, 0,
             NND, NND, 8, nullptr, nullptr, nullptr, inv8, 0);
        softmax_rows<<<NND, 256, 0, stream>>>(S32, S16, NND);
        gemm(S16, NND, Vh, 192, 0, nullptr, 0, at1h + h * 8, 64, 0, 0,
             NND, 8, NND, nullptr, nullptr, nullptr, 1.0f, 0);
    }
    gemm(at1h, 64, a1o_h, 64, 1, nullptr, 0, h64bh, 64, 0, 0,
         NND, 64, 64, nullptr, nullptr, nullptr, 1.0f, 0);

    // ===== encoder stage 1: Linear(64->40) + BN + ELU -> feat_x (f32 out + f16) =====
    gemm(h64bh, 64, e1w_h, 64, 1, out + OFX, 40, feat16, 40, 0, 0,
         NND, 40, 64, F(11), F(12), F(13), 1.0f, 1);

    // ===== ResGatedGraphConv stack =====
    auto rgconv = [&](const _Float16* cin, int l, float* aggOut) {
        int Di = din[l], Do = dou[l];
        gemm(cin, Di, rgw[l][0], Di, 1, kb, Do, nullptr, 0, 0, 0,
             NND, Do, Di, F(pb[l] + 1), nullptr, nullptr, 1.0f, 0);
        gemm(cin, Di, rgw[l][1], Di, 1, qb, Do, nullptr, 0, 0, 0,
             NND, Do, Di, F(pb[l] + 3), nullptr, nullptr, 1.0f, 0);
        gemm(cin, Di, rgw[l][2], Di, 1, vb, Do, nullptr, 0, 0, 0,
             NND, Do, Di, F(pb[l] + 5), nullptr, nullptr, 1.0f, 0);
        gemm(cin, Di, rgw[l][3], Di, 1, aggOut, Do, nullptr, 0, 0, 0,
             NND, Do, Di, F(pb[l] + 7), nullptr, nullptr, 1.0f, 0);
        int tot = NEDG * Do;
        rg_edges<<<(tot + 255) / 256, 256, 0, stream>>>(kb, qb, vb, esrc, edst,
                                                        aggOut, NEDG, Do);
    };
    auto bnelu = [&](const float* in, _Float16* o, const float* sc, const float* sh,
                     int D) {
        int tot = NND * D;
        bn_elu_h<<<(tot + 255) / 256, 256, 0, stream>>>(in, o, sc, sh, tot, D);
    };

    rgconv(feat16, 0, aggb);  bnelu(aggb, cb1h, F(22), F(23), 80);
    rgconv(cb1h,   1, aggb);  bnelu(aggb, cb2h, F(32), F(33), 64);
    rgconv(cb2h,   2, aggb);  bnelu(aggb, cb1h, F(42), F(43), 32);
    rgconv(cb1h,   3, aggb);  bnelu(aggb, cb2h, F(52), F(53), 16);
    rgconv(cb2h,   4, out + OMU);    // mu  (== gnn_z)
    rgconv(cb2h,   5, out + OLV);    // logvar

    // ===== z = [feat_x | mu] (f32), gnn_z = mu; z16 for the decoder =====
    copy_cols<<<(NND * 40 + 255) / 256, 256, 0, stream>>>(out + OZ, 48, out + OFX, 40, NND, 40);
    copy_cols<<<(NND * 8 + 255) / 256, 256, 0, stream>>>(out + OZ + 40, 48, out + OMU, 8, NND, 8);
    copy_cols<<<(NND * 8 + 255) / 256, 256, 0, stream>>>(out + OGZ, 8, out + OMU, 8, NND, 8);
    cvt(out + OZ, z16, NND * 48);

    // ===== decoder: d0 (48->64, BN+ELU) then d1 (64->3000) =====
    gemm(z16, 48, d0w_h, 48, 1, nullptr, 0, dbufh, 64, 0, 0,
         NND, 64, 48, F(71), F(72), F(73), 1.0f, 1);
    gemm(dbufh, 64, d1w_h, 64, 1, out + ODE, INDIM, nullptr, 0, 0, 0,
         NND, INDIM, 64, F(75), nullptr, nullptr, 1.0f, 0);

    // ===== Student-t cluster soft assignment =====
    cluster_q_k<<<(NND + 255) / 256, 256, 0, stream>>>(out + OZ, F(76), out + OQ, NND);
}
